// Qwen2VisionModel_35579509080177
// MI455X (gfx1250) — matrix operand order repaired and checked by reading, not compiled, not run on hardware
//
#include <hip/hip_runtime.h>
#include <hip/hip_bf16.h>
#include <math.h>

// ---------------------------------------------------------------------------
// Types for CDNA5 WMMA (gfx1250, wave32): D(f32 16x16) = A(bf16 16x32) x B(bf16 32x16) + C
// ---------------------------------------------------------------------------
typedef __bf16 bf16_t;
typedef bf16_t v16bf __attribute__((ext_vector_type(16)));
typedef float v8f __attribute__((ext_vector_type(8)));

union FragBF {
    v16bf v;
    unsigned u[8];
};

__device__ __forceinline__ unsigned short bf16_rne(float f) {
    unsigned x = __float_as_uint(f);
    x = (x + 0x7FFFu + ((x >> 16) & 1u)) >> 16;
    return (unsigned short)x;
}
__device__ __forceinline__ unsigned pk_bf16(float a, float b) {
    return (unsigned)bf16_rne(a) | ((unsigned)bf16_rne(b) << 16);
}
__device__ __forceinline__ v8f v8f_zero() {
    v8f r;
#pragma unroll
    for (int i = 0; i < 8; ++i) r[i] = 0.f;
    return r;
}
__device__ __forceinline__ v8f wmma_bf16(const FragBF& a, const FragBF& b, v8f c) {
    return __builtin_amdgcn_wmma_f32_16x16x32_bf16(false, a.v, false, b.v, (short)0, c,
                                                   false, false);
}

// lane ^ MASK exchange via ds_swizzle (group-of-32 mode: and=0x1F, or=0, xor=MASK)
template <int MASK>
__device__ __forceinline__ float swz_xor(float v) {
    return __int_as_float(
        __builtin_amdgcn_ds_swizzle(__float_as_int(v), (MASK << 10) | 0x1F));
}

// ---------------------------------------------------------------------------
// Templated GEMM:  C[M,N] = epilogue( A[M,K] @ Bw[N,K]^T + bias, resid )
// Block: 256 threads (8 waves) -> 128x128 tile; each wave 64x32 (4x2 WMMA tiles)
// A/B staged to LDS as bf16 (rows padded to 40 elems = 80B, conflict-free b128 reads)
// ---------------------------------------------------------------------------
constexpr int EPI_NONE = 0;        // plain store
constexpr int EPI_BIAS = 1;        // + bias
constexpr int EPI_BIAS_RESID = 2;  // + bias + residual
constexpr int EPI_QGELU = 3;       // + bias, quick_gelu
constexpr int EPI_GELU = 4;        // + bias, exact gelu

template <int EPI>
__global__ __launch_bounds__(256) void gemm_k(const float* __restrict__ A,
                                              const float* __restrict__ Bw,
                                              const float* __restrict__ bias,
                                              const float* resid, float* C, int M, int N,
                                              int K) {
    __shared__ unsigned short As[128 * 40];
    __shared__ unsigned short Bs[128 * 40];
    const int tid = threadIdx.x;
    const int lane = tid & 31;
    const int wave = tid >> 5;
    const int wm = wave & 1;        // 2 M-halves of 64
    const int wn = wave >> 1;       // 4 N-quarters of 32
    const int hl = lane >> 4, lc = lane & 15;
    const int mBase = blockIdx.y * 128, nBase = blockIdx.x * 128;

    v8f c[4][2];
#pragma unroll
    for (int i = 0; i < 4; ++i)
#pragma unroll
        for (int j = 0; j < 2; ++j) c[i][j] = v8f_zero();

    const int ksteps = (K + 31) / 32;
    for (int kt = 0; kt < ksteps; ++kt) {
        const int k0 = kt * 32;
        // ---- stage A/B tiles into LDS as bf16 (f32 -> bf16 RNE on the fly) ----
        for (int idx = tid; idx < 1024; idx += 256) {
            int r = idx >> 3;
            int c4 = (idx & 7) << 2;
            int gk = k0 + c4;
            int gr = mBase + r;
            float4 fa;
            if (gk + 3 < K && gr < M) {
                fa = *(const float4*)(A + (size_t)gr * K + gk);
            } else {
                fa.x = (gr < M && gk + 0 < K) ? A[(size_t)gr * K + gk + 0] : 0.f;
                fa.y = (gr < M && gk + 1 < K) ? A[(size_t)gr * K + gk + 1] : 0.f;
                fa.z = (gr < M && gk + 2 < K) ? A[(size_t)gr * K + gk + 2] : 0.f;
                fa.w = (gr < M && gk + 3 < K) ? A[(size_t)gr * K + gk + 3] : 0.f;
            }
            *(uint2*)(As + r * 40 + c4) =
                make_uint2(pk_bf16(fa.x, fa.y), pk_bf16(fa.z, fa.w));
            int gn = nBase + r;
            float4 fb;
            if (gk + 3 < K && gn < N) {
                fb = *(const float4*)(Bw + (size_t)gn * K + gk);
            } else {
                fb.x = (gn < N && gk + 0 < K) ? Bw[(size_t)gn * K + gk + 0] : 0.f;
                fb.y = (gn < N && gk + 1 < K) ? Bw[(size_t)gn * K + gk + 1] : 0.f;
                fb.z = (gn < N && gk + 2 < K) ? Bw[(size_t)gn * K + gk + 2] : 0.f;
                fb.w = (gn < N && gk + 3 < K) ? Bw[(size_t)gn * K + gk + 3] : 0.f;
            }
            *(uint2*)(Bs + r * 40 + c4) =
                make_uint2(pk_bf16(fb.x, fb.y), pk_bf16(fb.z, fb.w));
            // prefetch next K tile (global_prefetch_b8)
            if (gk + 32 < K && gr < M) __builtin_prefetch(A + (size_t)gr * K + gk + 32, 0, 1);
            if (gk + 32 < K && gn < N) __builtin_prefetch(Bw + (size_t)gn * K + gk + 32, 0, 1);
        }
        __syncthreads();

        // ---- fragments from LDS per documented bf16 layouts ----
        FragBF a[4], b[2];
#pragma unroll
        for (int i = 0; i < 4; ++i) {
            int r = wm * 64 + i * 16 + lc;
            int kc = hl * 8;  // A: lanes 0-15 hold K 0-7 & 16-23; lanes 16-31 K 8-15 & 24-31
            *(uint4*)&a[i].u[0] = *(const uint4*)(As + r * 40 + kc);
            *(uint4*)&a[i].u[4] = *(const uint4*)(As + r * 40 + kc + 16);
        }
#pragma unroll
        for (int j = 0; j < 2; ++j) {
            int n = wn * 32 + j * 16 + lc;
            int kc = hl * 8;  // B: lanes 0-15 hold K 0-7 & 16-23; lanes 16-31 K 8-15 & 24-31
            *(uint4*)&b[j].u[0] = *(const uint4*)(Bs + n * 40 + kc);
            *(uint4*)&b[j].u[4] = *(const uint4*)(Bs + n * 40 + kc + 16);
        }
#pragma unroll
        for (int i = 0; i < 4; ++i)
#pragma unroll
            for (int j = 0; j < 2; ++j) c[i][j] = wmma_bf16(a[i], b[j], c[i][j]);
        __syncthreads();
    }

    // ---- epilogue: C layout -> VGPR vi: lanes0-15 row=vi, lanes16-31 row=vi+8 ----
#pragma unroll
    for (int i = 0; i < 4; ++i) {
#pragma unroll
        for (int j = 0; j < 2; ++j) {
            int col = nBase + wn * 32 + j * 16 + lc;
            if (col >= N) continue;
            float bv = (EPI != EPI_NONE) ? bias[col] : 0.f;
#pragma unroll
            for (int vi = 0; vi < 8; ++vi) {
                int row = mBase + wm * 64 + i * 16 + vi + hl * 8;
                if (row >= M) continue;
                float v = c[i][j][vi] + bv;
                if (EPI == EPI_QGELU) {
                    v = v / (1.f + __expf(-1.702f * v));
                } else if (EPI == EPI_GELU) {
                    v = 0.5f * v * (1.f + erff(v * 0.7071067811865475f));
                }
                if (EPI == EPI_BIAS_RESID) v += resid[(size_t)row * N + col];
                C[(size_t)row * N + col] = v;
            }
        }
    }
}

// ---------------------------------------------------------------------------
// LayerNorm: one row per block, 256 threads
// ---------------------------------------------------------------------------
__global__ __launch_bounds__(256) void ln_k(const float* __restrict__ x,
                                            const float* __restrict__ g,
                                            const float* __restrict__ b,
                                            float* __restrict__ y, int C) {
    int row = blockIdx.x;
    const float* xr = x + (size_t)row * C;
    float s = 0.f, sq = 0.f;
    for (int c = threadIdx.x; c < C; c += 256) {
        float v = xr[c];
        s += v;
        sq += v * v;
    }
    __shared__ float ss[256], ssq[256];
    ss[threadIdx.x] = s;
    ssq[threadIdx.x] = sq;
    __syncthreads();
    for (int o = 128; o > 0; o >>= 1) {
        if (threadIdx.x < o) {
            ss[threadIdx.x] += ss[threadIdx.x + o];
            ssq[threadIdx.x] += ssq[threadIdx.x + o];
        }
        __syncthreads();
    }
    float mu = ss[0] / (float)C;
    float var = ssq[0] / (float)C - mu * mu;
    float rs = rsqrtf(var + 1e-6f);
    for (int c = threadIdx.x; c < C; c += 256)
        y[(size_t)row * C + c] = (xr[c] - mu) * rs * g[c] + b[c];
}

// ---------------------------------------------------------------------------
// Position / segment / rotary tables from grid_thw (MERGE=2 permutation)
// ---------------------------------------------------------------------------
__global__ void pos_k(const int* __restrict__ grid, int G, float* __restrict__ cosT,
                      float* __restrict__ sinT, int* __restrict__ seglo,
                      int* __restrict__ seghi, int L) {
    int l = blockIdx.x * blockDim.x + threadIdx.x;
    if (l >= L) return;
    int acc = 0, hp = 0, wp = 0, ss = 0, se = 1;
    for (int g = 0; g < G; ++g) {
        int t = grid[g * 3 + 0], h = grid[g * 3 + 1], w = grid[g * 3 + 2];
        int fr = h * w;
        int tot = t * fr;
        if (l < acc + tot) {
            int local = l - acc;
            int frame = local / fr;
            int inner = local - frame * fr;
            int iw = inner & 1;
            int ih = (inner >> 1) & 1;
            int bw = (inner >> 2) % (w >> 1);
            int bh = inner / (2 * w);
            hp = 2 * bh + ih;
            wp = 2 * bw + iw;
            ss = acc + frame * fr;
            se = ss + fr;
            break;
        }
        acc += tot;
    }
    seglo[l] = ss;
    seghi[l] = se;
    const float lnth = logf(10000.f);
#pragma unroll
    for (int j = 0; j < 20; ++j) {
        float invj = expf(-lnth * (float)j / 20.f);
        float fh = (float)hp * invj;
        float fw = (float)wp * invj;
        cosT[l * 40 + j] = cosf(fh);
        sinT[l * 40 + j] = sinf(fh);
        cosT[l * 40 + 20 + j] = cosf(fw);
        sinT[l * 40 + 20 + j] = sinf(fw);
    }
}

// ---------------------------------------------------------------------------
// RoPE applied in-place to q,k halves of qkv [L, 3840]
// ---------------------------------------------------------------------------
__global__ void rope_k(float* qkv, const float* __restrict__ cosT,
                       const float* __restrict__ sinT, int L) {
    int idx = blockIdx.x * blockDim.x + threadIdx.x;
    int total = L * 16 * 40;
    if (idx >= total) return;
    int l = idx / 640;
    int r = idx - l * 640;
    int hh = r / 40;
    int d = r - hh * 40;
    float cs = cosT[l * 40 + d];
    float sn = sinT[l * 40 + d];
    float* q = qkv + (size_t)l * 3840 + hh * 80;
    float q1 = q[d], q2 = q[d + 40];
    q[d] = q1 * cs - q2 * sn;
    q[d + 40] = q2 * cs + q1 * sn;
    float* k = q + 1280;
    float k1 = k[d], k2 = k[d + 40];
    k[d] = k1 * cs - k2 * sn;
    k[d + 40] = k2 * cs + k1 * sn;
}

// ---------------------------------------------------------------------------
// Flash attention: 1 wave per (head, 16-row q tile). HD=80 padded to 96.
// Keys iterated 32 at a time within the query's segment (block-diag mask).
// ---------------------------------------------------------------------------
__global__ __launch_bounds__(32) void attn_k(const float* __restrict__ qkv,
                                             const int* __restrict__ seglo,
                                             const int* __restrict__ seghi,
                                             float* __restrict__ out, int L) {
    (void)L;
    const int lane = threadIdx.x;
    const int hl = lane >> 4, lc = lane & 15;
    const int head = blockIdx.y;
    const int qbase = blockIdx.x * 16;
    const int hq = head * 80;
    const int segs = seglo[qbase];
    const int sege = seghi[qbase];
    const float scale = 0.1118033988749895f;  // 1/sqrt(80)

    __shared__ unsigned short Plds[16 * 40];

    // Q fragments (rope already applied): 3 k-steps of 32 (d 0..95, pad zero)
    FragBF qf[3];
    {
        const float* qrow = qkv + (size_t)(qbase + lc) * 3840 + hq;
#pragma unroll
        for (int s = 0; s < 3; ++s) {
            int k1 = s * 32 + hl * 16;
            if (k1 < 80) {
                float4 f0 = *(const float4*)(qrow + k1);
                float4 f1 = *(const float4*)(qrow + k1 + 4);
                qf[s].u[0] = pk_bf16(f0.x, f0.y);
                qf[s].u[1] = pk_bf16(f0.z, f0.w);
                qf[s].u[2] = pk_bf16(f1.x, f1.y);
                qf[s].u[3] = pk_bf16(f1.z, f1.w);
            } else {
                qf[s].u[0] = qf[s].u[1] = qf[s].u[2] = qf[s].u[3] = 0u;
            }
            int k2 = k1 + 8;
            if (k2 < 80) {
                float4 f0 = *(const float4*)(qrow + k2);
                float4 f1 = *(const float4*)(qrow + k2 + 4);
                qf[s].u[4] = pk_bf16(f0.x, f0.y);
                qf[s].u[5] = pk_bf16(f0.z, f0.w);
                qf[s].u[6] = pk_bf16(f1.x, f1.y);
                qf[s].u[7] = pk_bf16(f1.z, f1.w);
            } else {
                qf[s].u[4] = qf[s].u[5] = qf[s].u[6] = qf[s].u[7] = 0u;
            }
        }
    }

    v8f o[5];
#pragma unroll
    for (int dt = 0; dt < 5; ++dt) o[dt] = v8f_zero();
    float m_run[8], l_run[8];
#pragma unroll
    for (int vi = 0; vi < 8; ++vi) {
        m_run[vi] = -3.0e38f;
        l_run[vi] = 0.f;
    }

    for (int kb0 = segs; kb0 < sege; kb0 += 32) {
        v8f s0 = v8f_zero(), s1 = v8f_zero();
#pragma unroll
        for (int s = 0; s < 3; ++s) {
            int d0 = s * 32 + hl * 16;
#pragma unroll
            for (int t = 0; t < 2; ++t) {
                int keyn = kb0 + t * 16 + lc;
                int ka = keyn < sege ? keyn : sege - 1;
                const float* kr = qkv + (size_t)ka * 3840 + 1280 + hq;
                FragBF kf;
                if (d0 < 80) {
                    float4 f0 = *(const float4*)(kr + d0);
                    float4 f1 = *(const float4*)(kr + d0 + 4);
                    float4 f2 = *(const float4*)(kr + d0 + 8);
                    float4 f3 = *(const float4*)(kr + d0 + 12);
                    kf.u[0] = pk_bf16(f0.x, f0.y);
                    kf.u[1] = pk_bf16(f0.z, f0.w);
                    kf.u[2] = pk_bf16(f1.x, f1.y);
                    kf.u[3] = pk_bf16(f1.z, f1.w);
                    kf.u[4] = pk_bf16(f2.x, f2.y);
                    kf.u[5] = pk_bf16(f2.z, f2.w);
                    kf.u[6] = pk_bf16(f3.x, f3.y);
                    kf.u[7] = pk_bf16(f3.z, f3.w);
                } else {
#pragma unroll
                    for (int z = 0; z < 8; ++z) kf.u[z] = 0u;
                }
                if (t == 0)
                    s0 = wmma_bf16(qf[s], kf, s0);
                else
                    s1 = wmma_bf16(qf[s], kf, s1);
            }
        }

        // online softmax (row stats land one-per-lane per VGPR slot);
        // cross-lane reductions via ds_swizzle xor patterns (SWAPX1/2/4/8)
        bool ok0 = (kb0 + lc) < sege;
        bool ok1 = (kb0 + 16 + lc) < sege;
        float alpha_v[8];
#pragma unroll
        for (int vi = 0; vi < 8; ++vi) {
            float a0 = ok0 ? s0[vi] * scale : -1e9f;
            float a1 = ok1 ? s1[vi] * scale : -1e9f;
            float mx = fmaxf(a0, a1);
            mx = fmaxf(mx, swz_xor<1>(mx));
            mx = fmaxf(mx, swz_xor<2>(mx));
            mx = fmaxf(mx, swz_xor<4>(mx));
            mx = fmaxf(mx, swz_xor<8>(mx));
            float mn = fmaxf(m_run[vi], mx);
            float al = __expf(m_run[vi] - mn);
            float p0 = __expf(a0 - mn);
            float p1 = __expf(a1 - mn);
            float rs = p0 + p1;
            rs += swz_xor<1>(rs);
            rs += swz_xor<2>(rs);
            rs += swz_xor<4>(rs);
            rs += swz_xor<8>(rs);
            l_run[vi] = l_run[vi] * al + rs;
            m_run[vi] = mn;
            alpha_v[vi] = al;
            int prow = vi + hl * 8;
            Plds[prow * 40 + lc] = bf16_rne(p0);
            Plds[prow * 40 + 16 + lc] = bf16_rne(p1);
        }
        __syncthreads();  // single-wave WG: compiler fence + in-order DS

        FragBF pf;
        {
            int kc = hl * 16;
            *(uint4*)&pf.u[0] = *(const uint4*)(Plds + lc * 40 + kc);
            *(uint4*)&pf.u[4] = *(const uint4*)(Plds + lc * 40 + kc + 8);
        }

#pragma unroll
        for (int dt = 0; dt < 5; ++dt) {
#pragma unroll
            for (int vi = 0; vi < 8; ++vi) o[dt][vi] *= alpha_v[vi];
            FragBF vf;
            int dcol = dt * 16 + lc;
            int r0 = kb0 + hl * 16;
#pragma unroll
            for (int jj = 0; jj < 8; ++jj) {
                int ra = r0 + 2 * jj;
                int rb = ra + 1;
                ra = ra < sege ? ra : sege - 1;
                rb = rb < sege ? rb : sege - 1;
                float va = qkv[(size_t)ra * 3840 + 2560 + hq + dcol];
                float vb = qkv[(size_t)rb * 3840 + 2560 + hq + dcol];
                vf.u[jj] = pk_bf16(va, vb);
            }
            o[dt] = wmma_bf16(pf, vf, o[dt]);
        }
        __syncthreads();
    }

#pragma unroll
    for (int vi = 0; vi < 8; ++vi) {
        float inv = 1.f / l_run[vi];
        int row = qbase + vi + hl * 8;
#pragma unroll
        for (int dt = 0; dt < 5; ++dt)
            out[(size_t)row * 1280 + hq + dt * 16 + lc] = o[dt][vi] * inv;
    }
}

// ---------------------------------------------------------------------------
// Driver
// ---------------------------------------------------------------------------
extern "C" void kernel_launch(void* const* d_in, const int* in_sizes, int n_in,
                              void* d_out, int out_size, void* d_ws, size_t ws_size,
                              hipStream_t stream) {
    (void)n_in;
    (void)out_size;
    (void)ws_size;
    const float* pv = (const float*)d_in[0];
    const float* patchw = (const float*)d_in[1];
    const float* qkvw = (const float*)d_in[2];
    const float* qkvb = (const float*)d_in[3];
    const float* projw = (const float*)d_in[4];
    const float* projb = (const float*)d_in[5];
    const float* n1g = (const float*)d_in[6];
    const float* n1b = (const float*)d_in[7];
    const float* n2g = (const float*)d_in[8];
    const float* n2b = (const float*)d_in[9];
    const float* fc1w = (const float*)d_in[10];
    const float* fc1b = (const float*)d_in[11];
    const float* fc2w = (const float*)d_in[12];
    const float* fc2b = (const float*)d_in[13];
    const float* lnqg = (const float*)d_in[14];
    const float* lnqb = (const float*)d_in[15];
    const float* m1w = (const float*)d_in[16];
    const float* m1b = (const float*)d_in[17];
    const float* m2w = (const float*)d_in[18];
    const float* m2b = (const float*)d_in[19];
    const int* grid_thw = (const int*)d_in[20];
    const int L = in_sizes[0] / 1176;  // 2048
    const int G = in_sizes[20] / 3;

    // workspace layout (f32)
    float* x = (float*)d_ws;                     // [L,1280]
    float* h = x + (size_t)L * 1280;             // [L,1280]
    float* qkv = h + (size_t)L * 1280;           // [L,3840]
    float* attno = qkv + (size_t)L * 3840;       // [L,1280]
    float* mlp = attno + (size_t)L * 1280;       // [L,5120]
    float* y1 = mlp + (size_t)L * 5120;          // [L/4,5120]
    float* cosT = y1 + (size_t)(L / 4) * 5120;   // [L,40]
    float* sinT = cosT + (size_t)L * 40;         // [L,40]
    int* seglo = (int*)(sinT + (size_t)L * 40);  // [L]
    int* seghi = seglo + L;                      // [L]

    pos_k<<<(L + 255) / 256, 256, 0, stream>>>(grid_thw, G, cosT, sinT, seglo, seghi, L);

    // patch embed: x = pv @ patch_w^T   (K=1176 remainder handled)
    gemm_k<EPI_NONE><<<dim3(1280 / 128, L / 128), 256, 0, stream>>>(
        pv, patchw, nullptr, nullptr, x, L, 1280, 1176);

    for (int i = 0; i < 4; ++i) {
        ln_k<<<L, 256, 0, stream>>>(x, n1g + i * 1280, n1b + i * 1280, h, 1280);
        gemm_k<EPI_BIAS><<<dim3(3840 / 128, L / 128), 256, 0, stream>>>(
            h, qkvw + (size_t)i * 3840 * 1280, qkvb + i * 3840, nullptr, qkv, L, 3840, 1280);
        rope_k<<<(L * 640 + 255) / 256, 256, 0, stream>>>(qkv, cosT, sinT, L);
        attn_k<<<dim3(L / 16, 16), 32, 0, stream>>>(qkv, seglo, seghi, attno, L);
        gemm_k<EPI_BIAS_RESID><<<dim3(1280 / 128, L / 128), 256, 0, stream>>>(
            attno, projw + (size_t)i * 1280 * 1280, projb + i * 1280, x, x, L, 1280, 1280);
        ln_k<<<L, 256, 0, stream>>>(x, n2g + i * 1280, n2b + i * 1280, h, 1280);
        gemm_k<EPI_QGELU><<<dim3(5120 / 128, L / 128), 256, 0, stream>>>(
            h, fc1w + (size_t)i * 5120 * 1280, fc1b + i * 5120, nullptr, mlp, L, 5120, 1280);
        gemm_k<EPI_BIAS_RESID><<<dim3(1280 / 128, L / 128), 256, 0, stream>>>(
            mlp, fc2w + (size_t)i * 1280 * 5120, fc2b + i * 1280, x, x, L, 1280, 5120);
    }

    // merger: LN -> [L/4,5120] -> gelu(m1) -> m2
    ln_k<<<L, 256, 0, stream>>>(x, lnqg, lnqb, h, 1280);
    gemm_k<EPI_GELU><<<dim3(5120 / 128, (L / 4) / 128), 256, 0, stream>>>(
        h, m1w, m1b, nullptr, y1, L / 4, 5120, 5120);
    gemm_k<EPI_BIAS><<<dim3(3584 / 128, (L / 4) / 128), 256, 0, stream>>>(
        y1, m2w, m2b, nullptr, (float*)d_out, L / 4, 3584, 5120);
}
